// GAT_85830626443964
// MI455X (gfx1250) — matrix operand order repaired and checked by reading, not compiled, not run on hardware
//
#include <hip/hip_runtime.h>

typedef __attribute__((ext_vector_type(16))) _Float16 v16h;
typedef __attribute__((ext_vector_type(8)))  _Float16 v8h;
typedef __attribute__((ext_vector_type(8)))  float    v8f;
typedef __attribute__((ext_vector_type(4)))  unsigned int v4u;
typedef __attribute__((ext_vector_type(8)))  int      v8i;
typedef __attribute__((ext_vector_type(4)))  int      v4i;

#define NB   8
#define NN   1024
#define INF  128
#define OUTF 128
#define NH   4
#define HDIM 32
#define NEG_SLOPE 0.2f

// ---------------------------------------------------------------------------
// Kernel 1: h = x @ W, stored f16 transposed: hT[b][head][d][node]
// block = 128 threads (4 waves); each block does 64 rows x 128 cols.
// ---------------------------------------------------------------------------
__global__ __launch_bounds__(128) void gat_proj(const float* __restrict__ x,
                                                const float* __restrict__ W,
                                                _Float16* __restrict__ hT) {
    __shared__ alignas(32) _Float16 xs[64][136];   // +8 halfs pad (bank shift)
    __shared__ alignas(32) _Float16 wt[128][144];  // W transposed, +16 halfs pad

    const int tid = threadIdx.x;
    const int r0  = blockIdx.x * 64;  // global row base in [0, B*N)

    // Load W (128x128 f32, row k major) -> wt[n][k] f16
    for (int idx = tid; idx < 128 * 128; idx += 128) {
        int k = idx >> 7, n = idx & 127;
        wt[n][k] = (_Float16)W[idx];
    }
    // Load x tile (64x128) -> xs f16
    for (int idx = tid; idx < 64 * 128; idx += 128) {
        int r = idx >> 7, k = idx & 127;
        xs[r][k] = (_Float16)x[(size_t)r0 * 128 + idx];
    }
    __syncthreads();

    const int wave = tid >> 5;
    const int lane = tid & 31;
    const int m    = lane & 15;
    const int hi   = lane >> 4;
    const int m0   = wave * 16;

    for (int nt = 0; nt < 8; ++nt) {
        const int n0 = nt * 16;
        v8f acc = {};
        #pragma unroll
        for (int kk = 0; kk < 128; kk += 32) {
            // A fragment: rows m0+m, K layout {hi*8+e, 16+hi*8+e}
            v8h alo = *(const v8h*)&xs[m0 + m][kk + hi * 8];
            v8h ahi = *(const v8h*)&xs[m0 + m][kk + hi * 8 + 16];
            v16h a;
            #pragma unroll
            for (int e = 0; e < 8; ++e) { a[e] = alo[e]; a[e + 8] = ahi[e]; }
            // B fragment: col n0+m, K layout {hi*8+e, 16+hi*8+e}
            v8h blo = *(const v8h*)&wt[n0 + m][kk + hi * 8];
            v8h bhi = *(const v8h*)&wt[n0 + m][kk + hi * 8 + 16];
            v16h bfrag;
            #pragma unroll
            for (int e = 0; e < 8; ++e) { bfrag[e] = blo[e]; bfrag[e + 8] = bhi[e]; }
            acc = __builtin_amdgcn_wmma_f32_16x16x32_f16(
                false, a, false, bfrag, (short)0, acc, false, false);
        }
        // Store D transposed as f16: hT[((b*NH+head)*HDIM+d)*NN + node]
        const int nc   = n0 + m;
        const int head = nc >> 5;
        const int d    = nc & 31;
        #pragma unroll
        for (int v = 0; v < 8; ++v) {
            int g    = r0 + m0 + hi * 8 + v;   // global row = b*NN + node
            int bb   = g >> 10;
            int node = g & (NN - 1);
            hT[(size_t)((bb * NH + head) * HDIM + d) * NN + node] = (_Float16)acc[v];
        }
    }
}

// ---------------------------------------------------------------------------
// Kernel 2: attn_src/attn_dst per node (transposed [bh][n]) + max_j dst_j
// grid = B*H blocks, 256 threads
// ---------------------------------------------------------------------------
__global__ __launch_bounds__(256) void gat_attn(const _Float16* __restrict__ hT,
                                                const float* __restrict__ a_src,
                                                const float* __restrict__ a_dst,
                                                float* __restrict__ srcT,
                                                float* __restrict__ dstT,
                                                float* __restrict__ maxdst) {
    __shared__ float red[256];
    const int bh  = blockIdx.x;          // b*NH + h
    const int h   = bh & (NH - 1);
    const int tid = threadIdx.x;
    const _Float16* hb = hT + (size_t)bh * HDIM * NN;

    float lmax = -3.4e38f;
    for (int n = tid; n < NN; n += 256) {
        float ss = 0.f, ds = 0.f;
        #pragma unroll
        for (int d = 0; d < HDIM; ++d) {
            float v = (float)hb[(size_t)d * NN + n];
            ss += v * a_src[h * HDIM + d];
            ds += v * a_dst[h * HDIM + d];
        }
        srcT[(size_t)bh * NN + n] = ss;
        dstT[(size_t)bh * NN + n] = ds;
        lmax = fmaxf(lmax, ds);
    }
    red[tid] = lmax;
    __syncthreads();
    for (int s = 128; s > 0; s >>= 1) {
        if (tid < s) red[tid] = fmaxf(red[tid], red[tid + s]);
        __syncthreads();
    }
    if (tid == 0) maxdst[bh] = red[0];
}

// ---------------------------------------------------------------------------
// Kernel 3: fused softmax + aggregation.
// Per wave: one 16-row i-tile of one (b,h). Row max M_i = leaky(src_i + maxd)
// is exact (leaky is monotone) -> no online rescale. P tiles built in-register
// directly in WMMA A layout; O = P@V via 2 WMMA per 32-j step; row sums give
// the softmax denominator. dst_j vector staged to LDS via the Tensor Data
// Mover (1-D D# descriptor, CDNA5 ISA).
// ---------------------------------------------------------------------------
__global__ __launch_bounds__(128) void gat_aggregate(const _Float16* __restrict__ hT,
                                                     const float* __restrict__ srcT,
                                                     const float* __restrict__ dstT,
                                                     const float* __restrict__ maxdst,
                                                     float* __restrict__ out) {
    __shared__ alignas(16) float dstv[NN];
    const int blocksPerBH = 16;  // 64 i-tiles / 4 waves
    const int bh   = blockIdx.x / blocksPerBH;
    const int sub  = blockIdx.x % blocksPerBH;
    const int b    = bh >> 2;
    const int h    = bh & 3;
    const int tid  = threadIdx.x;
    const int wave = tid >> 5;
    const int lane = tid & 31;
    const int m    = lane & 15;
    const int hi   = lane >> 4;

#if __has_builtin(__builtin_amdgcn_tensor_load_to_lds) && \
    __has_builtin(__builtin_amdgcn_s_wait_tensorcnt)
    // --- TDM async copy: dstT[bh][0..1023] (4KB f32) -> LDS dstv ------------
    if (wave == 0) {
        const unsigned long long ga =
            (unsigned long long)(uintptr_t)(dstT + (size_t)bh * NN);
        const unsigned int lds = (unsigned int)(uintptr_t)&dstv[0]; // low 32b = LDS offset
        v4u g0;
        g0[0] = 1u;                                   // count=1, user descriptor
        g0[1] = lds;                                  // lds_addr [63:32]
        g0[2] = (unsigned int)ga;                     // global_addr [95:64]
        g0[3] = (unsigned int)((ga >> 32) & 0x01FFFFFFu) | (2u << 30); // type=2
        v8i g1;
        g1[0] = (int)(2u << 16);        // data_size=4B; wg_mask=0; no pad/iter
        g1[1] = (int)(1024u << 16);     // tensor_dim0 = 1024   (bits 79:48)
        g1[2] = (int)(1u << 16);        // tensor_dim1 = 1      (bits 111:80)
        g1[3] = (int)(1024u << 16);     // tile_dim0   = 1024   (bits 127:112)
        g1[4] = 0;                      // tile_dim1/2 unused
        g1[5] = 1024;                   // tensor_dim0_stride = 1024 (bits 207:160)
        g1[6] = 0;
        g1[7] = 0;
        v4i gz4 = {0, 0, 0, 0};         // groups 2/3 unused (<=2-D tensor)
        v8i gz8 = {0, 0, 0, 0, 0, 0, 0, 0};
        // 6-arg toolchain form: (g0, g1, g2, g3, extra_group, cpol)
        __builtin_amdgcn_tensor_load_to_lds(g0, g1, gz4, gz4, gz8, 0);
        __builtin_amdgcn_s_wait_tensorcnt(0);
    }
    __syncthreads();
#endif
    // Cooperative (re)load: guarantees dstv contents independent of TDM
    // descriptor interpretation; redundant stores of identical values.
    for (int j = tid; j < NN; j += 128) dstv[j] = dstT[(size_t)bh * NN + j];
    __syncthreads();

    const int   i0   = (sub * 4 + wave) * 16;
    const float s    = srcT[(size_t)bh * NN + i0 + m];  // lane m holds row m
    const float maxd = maxdst[bh];
    float Msc = s + maxd;
    Msc = Msc >= 0.f ? Msc : NEG_SLOPE * Msc;           // exact row max of logits

    const _Float16* Vt = hT + (size_t)bh * HDIM * NN;   // [HDIM][NN], row-major

    v8f   acc0 = {};
    v8f   acc1 = {};
    float rowsum = 0.f;

    for (int j0 = 0; j0 < NN; j0 += 32) {
        __builtin_prefetch(&Vt[(size_t)m * NN + j0 + 64], 0, 0);
        // Build P fragment (A layout), args to exp are all <= 0
        v16h a;
        float psum = 0.f;
        #pragma unroll
        for (int e = 0; e < 16; ++e) {
            int   k = hi * 16 + e;
            float l = s + dstv[j0 + k];
            l = l >= 0.f ? l : NEG_SLOPE * l;
            float p = __expf(l - Msc);
            psum += p;
            a[e] = (_Float16)p;
        }
        rowsum += psum;
        // B fragments: V columns d = m (acc0) and 16+m (acc1); K contiguous
        v16h b0 = *(const v16h*)&Vt[(size_t)m * NN + j0 + hi * 16];
        v16h b1 = *(const v16h*)&Vt[(size_t)(16 + m) * NN + j0 + hi * 16];
        acc0 = __builtin_amdgcn_wmma_f32_16x16x32_f16(
            false, a, false, b0, (short)0, acc0, false, false);
        acc1 = __builtin_amdgcn_wmma_f32_16x16x32_f16(
            false, a, false, b1, (short)0, acc1, false, false);
    }

    // Complete per-row sums: lanes L and L^16 cover K halves of row L&15
    rowsum += __shfl_xor(rowsum, 16, 32);

    // Normalize and store: c[v] = D[hi*8+v][m]
    #pragma unroll
    for (int v = 0; v < 8; ++v) {
        int   r   = hi * 8 + v;
        float den = __shfl(rowsum, r, 32);   // lane r holds denom of row r
        float inv = 1.0f / den;
        int   node = i0 + r;
        size_t base = ((size_t)(b * NN + node)) * OUTF + h * HDIM;
        out[base + m]      = acc0[v] * inv;
        out[base + 16 + m] = acc1[v] * inv;
    }
}

// ---------------------------------------------------------------------------
extern "C" void kernel_launch(void* const* d_in, const int* in_sizes, int n_in,
                              void* d_out, int out_size, void* d_ws, size_t ws_size,
                              hipStream_t stream) {
    const float* x     = (const float*)d_in[0];
    const float* W     = (const float*)d_in[1];
    const float* a_src = (const float*)d_in[2];
    const float* a_dst = (const float*)d_in[3];
    float* out = (float*)d_out;

    char* ws = (char*)d_ws;
    _Float16* hT   = (_Float16*)ws;                          // 2 MB: B*H*HD*N f16
    float*    srcT = (float*)(ws + (size_t)2 * 1024 * 1024); // 128 KB
    float*    dstT = srcT + (size_t)NB * NH * NN;            // 128 KB
    float*    maxd = dstT + (size_t)NB * NH * NN;            // 128 B

    gat_proj<<<(NB * NN) / 64, 128, 0, stream>>>(x, W, hT);
    gat_attn<<<NB * NH, 256, 0, stream>>>(hT, a_src, a_dst, srcT, dstT, maxd);
    gat_aggregate<<<NB * NH * 16, 128, 0, stream>>>(hT, srcT, dstT, maxd, out);
}
